// SelfAttention_61297773248696
// MI455X (gfx1250) — matrix operand order repaired and checked by reading, not compiled, not run on hardware
//
#include <hip/hip_runtime.h>

typedef _Float16 half_t;
typedef __attribute__((ext_vector_type(16))) _Float16 v16h;
typedef __attribute__((ext_vector_type(8)))  _Float16 v8h;
typedef __attribute__((ext_vector_type(4)))  _Float16 v4h;
typedef __attribute__((ext_vector_type(8)))  float    v8f;

typedef unsigned int u32;
typedef u32 u32x4 __attribute__((ext_vector_type(4)));
typedef int i32x4 __attribute__((ext_vector_type(4)));
typedef int i32x8 __attribute__((ext_vector_type(8)));

#define B_ 4
#define L_ 2048
#define E_ 1024
#define H_ 16
#define D_ 64

#if defined(__gfx1250__) && __has_builtin(__builtin_amdgcn_tensor_load_to_lds)
#define HAVE_TDM 1
#else
#define HAVE_TDM 0
#endif

union Frag16 { v16h v; v8h h[2]; };

static __device__ __forceinline__ v8f wmma_f16(v16h a, v16h b, v8f c) {
  // D = A(16x32 f16) * B(32x16 f16) + C(16x16 f32)
  return __builtin_amdgcn_wmma_f32_16x16x32_f16(false, a, false, b, (short)0, c,
                                                false, false);
}

#if HAVE_TDM
static __device__ __forceinline__ u32 lds_addr_of(const void* p) {
  // generic -> LDS addrspace; AS3 pointers are 32-bit LDS byte offsets
  return (u32)(size_t)(__attribute__((address_space(3))) const void*)p;
}

// 2D tile DMA: global (row-major, `pitch` elements per row) -> LDS, with
// optional LDS padding of (pad_amount_code+1) DWORDs every 2^(pic+1) DWORDs.
static __device__ __forceinline__ void tdm_load_2d(
    u32 lds_byte_addr, const void* gaddr, u32 tile_w, u32 tile_h, u32 pitch,
    u32 dsz_code /*0=1B,1=2B,2=4B,3=8B*/, u32 pad_interval_code,
    u32 pad_amount_code, bool pad_en) {
  const unsigned long long ga = (unsigned long long)(size_t)gaddr;
  u32x4 g0;
  g0.x = 1u;  // count=1, user descriptor, no gather
  g0.y = lds_byte_addr;
  g0.z = (u32)(ga & 0xffffffffu);                       // global_addr[31:0]
  g0.w = (u32)((ga >> 32) & 0x01ffffffu) | (2u << 30);  // addr[56:32] | type=2
  i32x8 g1;
  g1[0] = (int)((dsz_code << 16) | ((pad_en ? 1u : 0u) << 20) |
                (pad_interval_code << 22) | (pad_amount_code << 25));
  g1[1] = (int)(tile_w << 16);  // tensor_dim0[15:0] (OOB bound = tile)
  g1[2] = (int)(tile_h << 16);  // tensor_dim1[15:0]
  g1[3] = (int)(tile_w << 16);  // tile_dim0
  g1[4] = (int)tile_h;          // tile_dim1 (tile_dim2 = 0)
  g1[5] = (int)pitch;           // tensor_dim0_stride[31:0]
  g1[6] = 0;
  g1[7] = 0;
  const i32x4 z4 = {0, 0, 0, 0};
  const i32x8 z8 = {0, 0, 0, 0, 0, 0, 0, 0};
  // 6-arg form (clang-23 / therock-10.0 headers)
  __builtin_amdgcn_tensor_load_to_lds(g0, g1, z4, z4, z8, 0);
}
#endif  // HAVE_TDM

// ---------------------------------------------------------------------------
// fp32 -> f16 conversion pass (vectorized, grid-stride)
// ---------------------------------------------------------------------------
__global__ void cvt_f32_to_f16(const float* __restrict__ in,
                               half_t* __restrict__ out, int n4) {
  for (int i = blockIdx.x * blockDim.x + threadIdx.x; i < n4;
       i += gridDim.x * blockDim.x) {
    float4 f = ((const float4*)in)[i];
    v4h h = {(half_t)f.x, (half_t)f.y, (half_t)f.z, (half_t)f.w};
    ((v4h*)out)[i] = h;
  }
}

// ---------------------------------------------------------------------------
// C[M,1024] = A[M,1024] @ W[1024,1024]^T  (torch Linear: W rows = out cols)
// 128x128 block tile, 8 waves, each wave 32x64 (2x4 WMMA tiles).
// Double-buffered LDS tiles; staged by the Tensor Data Mover when available.
// Optional f32 output with fused bias (final projection).
// ---------------------------------------------------------------------------
__global__ __launch_bounds__(256) void gemm_xwt(
    const half_t* __restrict__ A, const half_t* __restrict__ W,
    half_t* __restrict__ outH, float* __restrict__ outF,
    const float* __restrict__ bias) {
  constexpr int KD = 1024;
  constexpr int ND = 1024;
  __shared__ __align__(16) half_t As[2][128][40];  // 32 -> 40 halves (pad 16B)
  __shared__ __align__(16) half_t Bs[2][128][40];

  const int tid = threadIdx.x;
  const int lane = tid & 31, wave = tid >> 5;
  const int lmod = lane & 15, lhalf = lane >> 4;
  const int wm = wave & 3;   // 4 waves along M -> 32 rows each
  const int wn = wave >> 2;  // 2 waves along N -> 64 cols each
  const int row0 = blockIdx.y * 128;
  const int col0 = blockIdx.x * 128;
  const int lrow = tid >> 1;  // 0..127 (fallback staging)
  const int lseg = tid & 1;

  v8f acc[2][4] = {};

  auto stage = [&](int buf, int k0) {
#if HAVE_TDM
    if (wave == 0) {
      // 128 rows x 32 halves, row pitch 1024 halves, pad 4 DW per 16 DW row
      tdm_load_2d(lds_addr_of(&As[buf][0][0]), A + (size_t)row0 * KD + k0,
                  32, 128, KD, /*dsz=*/1, /*pic=*/3, /*pac=*/3, true);
      tdm_load_2d(lds_addr_of(&Bs[buf][0][0]), W + (size_t)col0 * KD + k0,
                  32, 128, KD, /*dsz=*/1, /*pic=*/3, /*pac=*/3, true);
    }
#else
    const half_t* aSrc = A + (size_t)(row0 + lrow) * KD + k0 + lseg * 16;
    const half_t* bSrc = W + (size_t)(col0 + lrow) * KD + k0 + lseg * 16;
    *(uint4*)&As[buf][lrow][lseg * 16]     = *(const uint4*)(aSrc);
    *(uint4*)&As[buf][lrow][lseg * 16 + 8] = *(const uint4*)(aSrc + 8);
    *(uint4*)&Bs[buf][lrow][lseg * 16]     = *(const uint4*)(bSrc);
    *(uint4*)&Bs[buf][lrow][lseg * 16 + 8] = *(const uint4*)(bSrc + 8);
    if (k0 + 32 < KD) {  // WGP-scope prefetch of next K slab
      __builtin_prefetch(aSrc + 32, 0, 3);
      __builtin_prefetch(bSrc + 32, 0, 3);
    }
#endif
  };
  auto stage_wait = [&]() {
#if HAVE_TDM
    if (wave == 0) __builtin_amdgcn_s_wait_tensorcnt(0);
#endif
  };

  stage(0, 0);
  stage_wait();
  __syncthreads();

  for (int k0 = 0; k0 < KD; k0 += 32) {
    const int cur = (k0 >> 5) & 1;
    if (k0 + 32 < KD) stage(cur ^ 1, k0 + 32);  // DMA next tile during compute

    Frag16 af[2], bf[4];
#pragma unroll
    for (int mt = 0; mt < 2; mt++) {
      // A layout: lane<16 -> K[0..7],K[16..23]; lane>=16 -> K[8..15],K[24..31]
      const half_t* p = &As[cur][wm * 32 + mt * 16 + lmod][8 * lhalf];
      af[mt].h[0] = *(const v8h*)p;
      af[mt].h[1] = *(const v8h*)(p + 16);
    }
#pragma unroll
    for (int nt = 0; nt < 4; nt++) {
      // B layout: lane<16 -> K[0..7],K[16..23]; lane>=16 -> K[8..15],K[24..31]
      const half_t* p = &Bs[cur][wn * 64 + nt * 16 + lmod][8 * lhalf];
      bf[nt].h[0] = *(const v8h*)p;
      bf[nt].h[1] = *(const v8h*)(p + 16);
    }
#pragma unroll
    for (int mt = 0; mt < 2; mt++)
#pragma unroll
      for (int nt = 0; nt < 4; nt++)
        acc[mt][nt] = wmma_f16(af[mt].v, bf[nt].v, acc[mt][nt]);

    stage_wait();
    __syncthreads();
  }

  // C/D layout: VGPR r, lanes 0-15 -> M=r, N=lane; lanes 16-31 -> M=r+8
#pragma unroll
  for (int mt = 0; mt < 2; mt++)
#pragma unroll
    for (int nt = 0; nt < 4; nt++)
#pragma unroll
      for (int r = 0; r < 8; r++) {
        const int row = row0 + wm * 32 + mt * 16 + r + 8 * lhalf;
        const int col = col0 + wn * 64 + nt * 16 + lmod;
        const float vv = acc[mt][nt][r];
        if (outF)
          outF[(size_t)row * ND + col] = vv + (bias ? bias[col] : 0.0f);
        else
          outH[(size_t)row * ND + col] = (half_t)vv;
      }
}

// ---------------------------------------------------------------------------
// Flash attention: one block = (b, h, 64 query rows). 4 waves, 16 rows each.
// S = QK^T via WMMA, masked-fill(-1e20) then *1/32, online softmax, O += P.V.
// K tile and mask tile staged by TDM; V staged manually (needs transpose).
// ---------------------------------------------------------------------------
__global__ __launch_bounds__(128) void flash_attn(
    const half_t* __restrict__ Q, const half_t* __restrict__ K,
    const half_t* __restrict__ V, const int* __restrict__ MSK,
    half_t* __restrict__ O) {
  __shared__ __align__(16) half_t Kt[64][72];     // [key][d], pad 128->144B
  __shared__ __align__(16) half_t Vt[64][72];     // transposed: [d][key]
  __shared__ __align__(16) int    Mk[64][64];     // mask tile [qrow][kcol]
  __shared__ __align__(16) half_t Pb[4][16][72];  // per-wave P staging

  const int tid = threadIdx.x;
  const int lane = tid & 31, wave = tid >> 5;
  const int lmod = lane & 15, lhalf = lane >> 4;
  const int b = blockIdx.z, h = blockIdx.y;
  const int q0 = blockIdx.x * 64;
  const int qw = q0 + wave * 16;  // this wave's first query row

  // Q A-fragments (16 rows x 64 d) live in registers across the whole loop
  Frag16 aq[2];
  {
    const half_t* qp = Q + (size_t)(b * L_ + qw + lmod) * E_ + h * D_;
#pragma unroll
    for (int j = 0; j < 2; j++) {
      aq[j].h[0] = *(const v8h*)(qp + j * 32 + 8 * lhalf);
      aq[j].h[1] = *(const v8h*)(qp + j * 32 + 8 * lhalf + 16);
    }
  }

  float mi[8], li[8];
  v8f acc[4] = {};
#pragma unroll
  for (int r = 0; r < 8; r++) {
    mi[r] = -3.0e38f;
    li[r] = 0.0f;
  }

  const int lrow = tid >> 1;  // 0..63 (key row / mask row)
  const int lseg = tid & 1;   // 32-element segment

  for (int k0 = 0; k0 < L_; k0 += 64) {
    // ---- stage K (TDM), mask (TDM), V^T (manual, overlaps the DMA) ----
#if HAVE_TDM
    if (wave == 0) {
      // K: 64 x 64 halves, pitch E_, pad 4 DW per 32 DW row (-> 72-half rows)
      tdm_load_2d(lds_addr_of(&Kt[0][0]),
                  K + (size_t)(b * L_ + k0) * E_ + h * D_, 64, 64, E_,
                  /*dsz=*/1, /*pic=*/4, /*pac=*/3, true);
      // mask: 64 x 64 ints, pitch L_, no padding
      tdm_load_2d(lds_addr_of(&Mk[0][0]),
                  MSK + (size_t)(b * L_ + q0) * L_ + k0, 64, 64, L_,
                  /*dsz=*/2, 0, 0, false);
    }
#else
    {
      const half_t* ks =
          K + (size_t)(b * L_ + k0 + lrow) * E_ + h * D_ + lseg * 32;
#pragma unroll
      for (int i = 0; i < 4; i++)
        *(uint4*)&Kt[lrow][lseg * 32 + i * 8] = *(const uint4*)(ks + i * 8);
      const int* ms = MSK + (size_t)(b * L_ + q0 + lrow) * L_ + k0 + lseg * 32;
#pragma unroll
      for (int i = 0; i < 8; i++)
        *(uint4*)&Mk[lrow][lseg * 32 + i * 4] = *(const uint4*)(ms + i * 4);
    }
#endif
    {
      __align__(16) half_t vbuf[32];
      const half_t* vs =
          V + (size_t)(b * L_ + k0 + lrow) * E_ + h * D_ + lseg * 32;
#pragma unroll
      for (int i = 0; i < 4; i++)
        *(uint4*)&vbuf[i * 8] = *(const uint4*)(vs + i * 8);
#pragma unroll
      for (int i = 0; i < 32; i++) Vt[lseg * 32 + i][lrow] = vbuf[i];
    }
#if HAVE_TDM
    if (wave == 0) __builtin_amdgcn_s_wait_tensorcnt(0);
#endif
    __syncthreads();

    // ---- S = Q K^T  (M=16 rows, N=64 keys, K=64) ----
    v8f sv[4] = {};
#pragma unroll
    for (int j = 0; j < 2; j++) {
#pragma unroll
      for (int n = 0; n < 4; n++) {
        Frag16 bk;
        const half_t* kp = &Kt[n * 16 + lmod][j * 32 + 8 * lhalf];
        bk.h[0] = *(const v8h*)kp;
        bk.h[1] = *(const v8h*)(kp + 16);
        sv[n] = wmma_f16(aq[j].v, bk.v, sv[n]);
      }
    }

    // ---- masked-fill before scaling (reference semantics), online softmax ----
    float p[4][8], mt[8];
#pragma unroll
    for (int r = 0; r < 8; r++) mt[r] = -3.0e38f;
#pragma unroll
    for (int n = 0; n < 4; n++)
#pragma unroll
      for (int r = 0; r < 8; r++) {
        float vv = sv[n][r];
        if (Mk[wave * 16 + r + 8 * lhalf][n * 16 + lmod] == 0) vv = -1.0e20f;
        vv *= 0.03125f;  // 1/sqrt(E) = 1/32
        p[n][r] = vv;
        mt[r] = fmaxf(mt[r], vv);
      }
    // rows are replicated across the 16 lanes of each half -> xor-reduce
#pragma unroll
    for (int off = 1; off < 16; off <<= 1)
#pragma unroll
      for (int r = 0; r < 8; r++)
        mt[r] = fmaxf(mt[r], __shfl_xor(mt[r], off, 32));

    float alpha[8], rs[8];
#pragma unroll
    for (int r = 0; r < 8; r++) {
      const float mn = fmaxf(mi[r], mt[r]);
      alpha[r] = __expf(mi[r] - mn);
      mi[r] = mn;
      rs[r] = 0.0f;
    }
#pragma unroll
    for (int n = 0; n < 4; n++)
#pragma unroll
      for (int r = 0; r < 8; r++) {
        const float e = __expf(p[n][r] - mi[r]);
        p[n][r] = e;
        rs[r] += e;
      }
#pragma unroll
    for (int off = 1; off < 16; off <<= 1)
#pragma unroll
      for (int r = 0; r < 8; r++) rs[r] += __shfl_xor(rs[r], off, 32);
#pragma unroll
    for (int r = 0; r < 8; r++) li[r] = li[r] * alpha[r] + rs[r];
#pragma unroll
    for (int n = 0; n < 4; n++)
#pragma unroll
      for (int r = 0; r < 8; r++) acc[n][r] *= alpha[r];

    // ---- C-layout -> A-layout for P via per-wave LDS region ----
#pragma unroll
    for (int n = 0; n < 4; n++)
#pragma unroll
      for (int r = 0; r < 8; r++)
        Pb[wave][r + 8 * lhalf][n * 16 + lmod] = (half_t)p[n][r];
    asm volatile("s_wait_dscnt 0" ::: "memory");

    // ---- O += P V  (M=16 rows, N=64 d, K=64 keys; B-frags from Vt rows) ----
#pragma unroll
    for (int j = 0; j < 2; j++) {
      Frag16 ap;
      const half_t* pp = &Pb[wave][lmod][j * 32 + 8 * lhalf];
      ap.h[0] = *(const v8h*)pp;
      ap.h[1] = *(const v8h*)(pp + 16);
#pragma unroll
      for (int n = 0; n < 4; n++) {
        Frag16 bv;
        const half_t* vp = &Vt[n * 16 + lmod][j * 32 + 8 * lhalf];
        bv.h[0] = *(const v8h*)vp;
        bv.h[1] = *(const v8h*)(vp + 16);
        acc[n] = wmma_f16(ap.v, bv.v, acc[n]);
      }
    }
    __syncthreads();
  }

  // ---- normalize and emit f16 attention output [B, L, H*D] ----
#pragma unroll
  for (int r = 0; r < 8; r++) {
    const float inv = 1.0f / li[r];
    const int row = qw + r + 8 * lhalf;
    half_t* op = O + (size_t)(b * L_ + row) * E_ + h * D_;
#pragma unroll
    for (int n = 0; n < 4; n++)
      op[n * 16 + lmod] = (half_t)(acc[n][r] * inv);
  }
}

// ---------------------------------------------------------------------------
extern "C" void kernel_launch(void* const* d_in, const int* in_sizes, int n_in,
                              void* d_out, int out_size, void* d_ws,
                              size_t ws_size, hipStream_t stream) {
  const float* queries = (const float*)d_in[0];
  const float* keys    = (const float*)d_in[1];
  const float* values  = (const float*)d_in[2];
  const int*   mask    = (const int*)d_in[3];
  const float* Wq = (const float*)d_in[4];
  const float* Wk = (const float*)d_in[5];
  const float* Wv = (const float*)d_in[6];
  const float* Wo = (const float*)d_in[7];
  const float* bo = (const float*)d_in[8];

  const size_t SZX = (size_t)B_ * L_ * E_;  // 8,388,608
  const size_t SZW = (size_t)E_ * E_;       // 1,048,576

  half_t* p  = (half_t*)d_ws;
  half_t* xq = p; p += SZX;
  half_t* xk = p; p += SZX;
  half_t* xv = p; p += SZX;
  half_t* wq = p; p += SZW;
  half_t* wk = p; p += SZW;
  half_t* wv = p; p += SZW;
  half_t* wo = p; p += SZW;
  half_t* qh = p; p += SZX;
  half_t* kh = p; p += SZX;
  half_t* vh = p; p += SZX;
  half_t* oh = p; p += SZX;  // total 120 MB of workspace

  auto cvt = [&](const float* src, half_t* dst, size_t n) {
    const int n4 = (int)(n / 4);
    int blocks = (n4 + 255) / 256;
    if (blocks > 8192) blocks = 8192;
    cvt_f32_to_f16<<<dim3(blocks), dim3(256), 0, stream>>>(src, dst, n4);
  };
  cvt(queries, xq, SZX);
  cvt(keys,    xk, SZX);
  cvt(values,  xv, SZX);
  cvt(Wq, wq, SZW);
  cvt(Wk, wk, SZW);
  cvt(Wv, wv, SZW);
  cvt(Wo, wo, SZW);

  const dim3 ggrid(1024 / 128, (B_ * L_) / 128);  // (8, 64)
  gemm_xwt<<<ggrid, 256, 0, stream>>>(xq, wq, qh, nullptr, nullptr);
  gemm_xwt<<<ggrid, 256, 0, stream>>>(xk, wk, kh, nullptr, nullptr);
  gemm_xwt<<<ggrid, 256, 0, stream>>>(xv, wv, vh, nullptr, nullptr);

  flash_attn<<<dim3(L_ / 64, H_, B_), 128, 0, stream>>>(qh, kh, vh, mask, oh);

  gemm_xwt<<<ggrid, 256, 0, stream>>>(oh, wo, nullptr, (float*)d_out, bo);
}
